// HierarchicalGraphEncoder_82918638617237
// MI455X (gfx1250) — matrix operand order repaired and checked by reading, not compiled, not run on hardware
//
#include <hip/hip_runtime.h>
#include <math.h>

// ---------------------------------------------------------------------------
// HierarchicalGraphEncoder for MI455X (gfx1250, wave32, WMMA).
// All GEMMs run through one NT-form WMMA kernel using v_wmma_f32_16x16x32_f16.
// S (N*N f32, 37.7MB) and P (N*N f16, 18.9MB) are L2-resident on MI455X
// (192MB L2), so dense materialization is bandwidth-optimal without fusion.
// ---------------------------------------------------------------------------

#define N_NODES 3072
#define EDGES   49152
#define DIM     256
#define OUT_D   512
#define HEADS   8
#define DHEAD   32
#define NET_T   5
#define L_LOC   2
#define L_GLB   2

typedef __attribute__((ext_vector_type(16))) _Float16 v16h;
typedef __attribute__((ext_vector_type(8)))  float    v8f;

union F16Frag { v16h v; uint4 u[2]; };
union AccFrag { v8f  v; float f[8]; };

// ---------------------------------------------------------------------------
// NT GEMM: C[M,Ncol] = alpha * A[M,K] * B[Ncol,K]^T (+ bias[Ncol])
// A, B row-major f16. Optional outputs: f32 row-major, f16 row-major,
// f16 transposed (Ct[n*ldt+m], used to produce V^T for the P@V GEMM).
// One wave -> one 16x16 tile. Workgroup tile = 64 rows x 32 cols (8 waves).
// Fragment layouts follow the CDNA5 ISA 16-bit A/B VGPR tables:
//   A: lanes 0-15 hold K (0..7, 16..23); lanes 16-31 hold K (8..15, 24..31)
//   B: lanes 0-15 hold K (0..7, 16..23); lanes 16-31 hold K (8..15, 24..31) (col n = lane%16)
// -> two 16B global loads per fragment per lane.
// ---------------------------------------------------------------------------
__global__ __launch_bounds__(256)
void gemm_nt_wmma(const _Float16* __restrict__ A, int lda,
                  const _Float16* __restrict__ B, int ldb,
                  int M, int Ncol, int Klen, float alpha,
                  const float* __restrict__ bias,
                  float*    __restrict__ Cf, int ldc,
                  _Float16* __restrict__ Ch, int ldh,
                  _Float16* __restrict__ Ct, int ldt)
{
    const int lane = threadIdx.x & 31;
    const int wave = threadIdx.x >> 5;
    const int hseg = lane >> 4;     // 0: lanes 0-15, 1: lanes 16-31
    const int lrow = lane & 15;
    const int m0 = blockIdx.y * 64 + (wave >> 1) * 16;
    const int n0 = blockIdx.x * 32 + (wave & 1) * 16;
    if (m0 >= M || n0 >= Ncol) return;

    const _Float16* arow = A + (size_t)(m0 + lrow) * lda;
    const _Float16* brow = B + (size_t)(n0 + lrow) * ldb;
    const int aoff = hseg * 8;    // A: K base 0 or 8, second chunk +16
    const int boff = hseg * 8;    // B: K base 0 or 8, second chunk +16

    v8f acc = {};
    for (int k = 0; k < Klen; k += 32) {
        F16Frag a, b;
        a.u[0] = *reinterpret_cast<const uint4*>(arow + k + aoff);
        a.u[1] = *reinterpret_cast<const uint4*>(arow + k + aoff + 16);
        b.u[0] = *reinterpret_cast<const uint4*>(brow + k + boff);
        b.u[1] = *reinterpret_cast<const uint4*>(brow + k + boff + 16);
        acc = __builtin_amdgcn_wmma_f32_16x16x32_f16(
                  false, a.v, false, b.v, (short)0, acc, false, false);
    }

    AccFrag r; r.v = acc;
    const int n = n0 + lrow;
    const float badd = bias ? bias[n] : 0.0f;
    #pragma unroll
    for (int i = 0; i < 8; ++i) {
        const int m = m0 + i + hseg * 8;   // C layout: VGPR i -> M=i / i+8
        const float val = r.f[i] * alpha + badd;
        if (Cf) Cf[(size_t)m * ldc + n] = val;
        if (Ch) Ch[(size_t)m * ldh + n] = (_Float16)val;
        if (Ct) Ct[(size_t)n * ldt + m] = (_Float16)val;
    }
}

// h[n,d] = x[n] * emb_w[d] + emb_b[d]
__global__ __launch_bounds__(256)
void embed_kernel(const float* __restrict__ x, const float* __restrict__ ew,
                  const float* __restrict__ eb, float* __restrict__ h)
{
    const int n = blockIdx.x, t = threadIdx.x;
    h[(size_t)n * DIM + t] = x[n] * ew[t] + eb[t];
}

// y16[i] = (f16)(x[i] + (pos ? pos[i] : 0))
__global__ __launch_bounds__(256)
void cvt_f16_kernel(const float* __restrict__ x, const float* __restrict__ pos,
                    _Float16* __restrict__ y, int count)
{
    const int i = blockIdx.x * 256 + threadIdx.x;
    if (i >= count) return;
    float v = x[i];
    if (pos) v += pos[i];
    y[i] = (_Float16)v;
}

// Wt16[o*Din + i] = (f16) W[i*Dout + o]
__global__ __launch_bounds__(256)
void wtrans_kernel(const float* __restrict__ W, _Float16* __restrict__ Wt,
                   int Din, int Dout)
{
    const int idx = blockIdx.x * 256 + threadIdx.x;
    if (idx >= Din * Dout) return;
    const int i = idx / Dout, o = idx % Dout;
    Wt[(size_t)o * Din + i] = (_Float16)W[idx];
}

// S[src, dst] += eb_layer[edge_type, head]   (L2-resident atomic scatter)
__global__ __launch_bounds__(256)
void scatter_edge_bias(float* __restrict__ S, const int* __restrict__ ei,
                       const int* __restrict__ et, const float* __restrict__ ebl,
                       int head)
{
    const int e = blockIdx.x * 256 + threadIdx.x;
    if (e >= EDGES) return;
    const int s = ei[e];
    const int d = ei[EDGES + e];
    atomicAdd(&S[(size_t)s * N_NODES + d], ebl[et[e] * HEADS + head]);
}

__device__ __forceinline__ float blk_sum(float v, float* red, int t)
{
    red[t] = v; __syncthreads();
    for (int s = 128; s > 0; s >>= 1) { if (t < s) red[t] += red[t + s]; __syncthreads(); }
    const float r = red[0]; __syncthreads();
    return r;
}
__device__ __forceinline__ float blk_max(float v, float* red, int t)
{
    red[t] = v; __syncthreads();
    for (int s = 128; s > 0; s >>= 1) { if (t < s) red[t] = fmaxf(red[t], red[t + s]); __syncthreads(); }
    const float r = red[0]; __syncthreads();
    return r;
}

// masked softmax row-wise: mode 0 keeps same-community cols, mode 1 different.
__global__ __launch_bounds__(256)
void masked_softmax(const float* __restrict__ S, _Float16* __restrict__ P,
                    const int* __restrict__ cid, int mode)
{
    __shared__ float red[256];
    const int row = blockIdx.x, t = threadIdx.x;
    const int myc = cid[row];
    const float* srow = S + (size_t)row * N_NODES;
    _Float16*    prow = P + (size_t)row * N_NODES;

    float mx = -INFINITY;
    for (int m = t; m < N_NODES; m += 256) {
        const bool keep = (cid[m] == myc) == (mode == 0);
        if (keep) mx = fmaxf(mx, srow[m]);
    }
    mx = blk_max(mx, red, t);

    float sum = 0.0f;
    for (int m = t; m < N_NODES; m += 256) {
        const bool keep = (cid[m] == myc) == (mode == 0);
        if (keep) sum += __expf(srow[m] - mx);
    }
    sum = blk_sum(sum, red, t);
    const float rinv = 1.0f / sum;

    for (int m = t; m < N_NODES; m += 256) {
        const bool keep = (cid[m] == myc) == (mode == 0);
        prow[m] = keep ? (_Float16)(__expf(srow[m] - mx) * rinv) : (_Float16)0.0f;
    }
}

// h = LayerNorm(h + o) * g + b    (one row per block, D == blockDim)
__global__ __launch_bounds__(256)
void residual_layernorm(float* __restrict__ h, const float* __restrict__ o,
                        const float* __restrict__ g, const float* __restrict__ b)
{
    __shared__ float red[256];
    const int row = blockIdx.x, t = threadIdx.x;
    const size_t idx = (size_t)row * DIM + t;
    const float v = h[idx] + o[idx];
    const float mean = blk_sum(v, red, t) * (1.0f / DIM);
    const float d = v - mean;
    const float var = blk_sum(d * d, red, t) * (1.0f / DIM);
    h[idx] = d * rsqrtf(var + 1e-5f) * g[t] + b[t];
}

// ---------------------------------------------------------------------------
extern "C" void kernel_launch(void* const* d_in, const int* in_sizes, int n_in,
                              void* d_out, int out_size, void* d_ws, size_t ws_size,
                              hipStream_t stream)
{
    (void)in_sizes; (void)n_in; (void)out_size; (void)ws_size;
    const float* x        = (const float*)d_in[0];
    const int*   ei       = (const int*)  d_in[1];
    const int*   et       = (const int*)  d_in[2];
    const float* pos      = (const float*)d_in[3];
    const int*   cid      = (const int*)  d_in[4];
    /* d_in[5] adj_matrix unused */
    const float* emb_w    = (const float*)d_in[6];
    const float* emb_b    = (const float*)d_in[7];
    const float* loc_qw   = (const float*)d_in[8];
    const float* loc_qb   = (const float*)d_in[9];
    const float* loc_kw   = (const float*)d_in[10];
    const float* loc_kb   = (const float*)d_in[11];
    const float* loc_vw   = (const float*)d_in[12];
    const float* loc_vb   = (const float*)d_in[13];
    const float* loc_ow   = (const float*)d_in[14];
    const float* loc_ob   = (const float*)d_in[15];
    const float* loc_eb   = (const float*)d_in[16];
    const float* loc_ln_g = (const float*)d_in[17];
    const float* loc_ln_b = (const float*)d_in[18];
    const float* glob_qw  = (const float*)d_in[19];
    const float* glob_qb  = (const float*)d_in[20];
    const float* glob_kw  = (const float*)d_in[21];
    const float* glob_kb  = (const float*)d_in[22];
    const float* glob_vw  = (const float*)d_in[23];
    const float* glob_vb  = (const float*)d_in[24];
    const float* glob_ow  = (const float*)d_in[25];
    const float* glob_ob  = (const float*)d_in[26];
    const float* glob_ln_g= (const float*)d_in[27];
    const float* glob_ln_b= (const float*)d_in[28];
    const float* out_w    = (const float*)d_in[29];
    const float* out_b    = (const float*)d_in[30];
    float* out = (float*)d_out;

    // ---- workspace carve-up (~73 MB total) ----
    char* ws = (char*)d_ws;
    size_t cur = 0;
    auto carve = [&](size_t bytes) -> char* {
        char* p = ws + cur;
        cur = (cur + bytes + 255) & ~(size_t)255;
        return p;
    };
    const size_t ND = (size_t)N_NODES * DIM;
    const size_t NN = (size_t)N_NODES * N_NODES;
    float*    h      = (float*)   carve(ND * 4);
    float*    tmp    = (float*)   carve(ND * 4);   // o-projection output
    float*    tmp2   = (float*)   carve(ND * 4);   // attention (P@V) output
    _Float16* a16    = (_Float16*)carve(ND * 2);   // f16 activations (hp or h)
    _Float16* q16    = (_Float16*)carve(ND * 2);
    _Float16* k16    = (_Float16*)carve(ND * 2);
    _Float16* att16  = (_Float16*)carve(ND * 2);
    _Float16* vt16   = (_Float16*)carve(ND * 2);   // V^T : [DIM rows, N cols]
    float*    S      = (float*)   carve(NN * 4);   // scores (L2-resident)
    _Float16* P16    = (_Float16*)carve(NN * 2);   // probs f16
    // transposed f16 weights: 2x4 local + 2x4 global (D*D) + out (O*D)
    _Float16* wT[16];
    for (int i = 0; i < 16; ++i) wT[i] = (_Float16*)carve((size_t)DIM * DIM * 2);
    _Float16* woutT = (_Float16*)carve((size_t)OUT_D * DIM * 2);

    const dim3 B256(256);
    auto wtrans = [&](const float* w, _Float16* dst, int Din, int Dout) {
        hipLaunchKernelGGL(wtrans_kernel, dim3((Din * Dout + 255) / 256), B256, 0,
                           stream, w, dst, Din, Dout);
    };
    auto gemm = [&](const _Float16* A, int lda, const _Float16* Bm, int ldb,
                    int M, int Nc, int K, float alpha, const float* bias,
                    float* Cf, int ldc, _Float16* Ch, int ldh,
                    _Float16* Ct, int ldt) {
        hipLaunchKernelGGL(gemm_nt_wmma, dim3(Nc / 32, M / 64), B256, 0, stream,
                           A, lda, Bm, ldb, M, Nc, K, alpha, bias,
                           Cf, ldc, Ch, ldh, Ct, ldt);
    };
    const int DD = DIM * DIM;
    const float inv_sqrt_dh = 0.17677669529663687f;   // 1/sqrt(32)

    // ---- weight preparation (transpose + f32->f16) ----
    for (int l = 0; l < L_LOC; ++l) {
        wtrans(loc_qw + l * DD, wT[l * 4 + 0], DIM, DIM);
        wtrans(loc_kw + l * DD, wT[l * 4 + 1], DIM, DIM);
        wtrans(loc_vw + l * DD, wT[l * 4 + 2], DIM, DIM);
        wtrans(loc_ow + l * DD, wT[l * 4 + 3], DIM, DIM);
    }
    for (int g = 0; g < L_GLB; ++g) {
        wtrans(glob_qw + g * DD, wT[8 + g * 4 + 0], DIM, DIM);
        wtrans(glob_kw + g * DD, wT[8 + g * 4 + 1], DIM, DIM);
        wtrans(glob_vw + g * DD, wT[8 + g * 4 + 2], DIM, DIM);
        wtrans(glob_ow + g * DD, wT[8 + g * 4 + 3], DIM, DIM);
    }
    wtrans(out_w, woutT, DIM, OUT_D);

    // ---- embedding ----
    hipLaunchKernelGGL(embed_kernel, dim3(N_NODES), B256, 0, stream, x, emb_w, emb_b, h);

    // ---- local (Graphormer) layers ----
    for (int l = 0; l < L_LOC; ++l) {
        // hp = h + pos  ->  f16
        hipLaunchKernelGGL(cvt_f16_kernel, dim3((int)(ND / 256)), B256, 0, stream,
                           h, pos, a16, (int)ND);
        // Q, K row-major f16; V written transposed (V^T [DIM, N])
        gemm(a16, DIM, wT[l*4+0], DIM, N_NODES, DIM, DIM, 1.0f, loc_qb + l*DIM,
             nullptr, 0, q16, DIM, nullptr, 0);
        gemm(a16, DIM, wT[l*4+1], DIM, N_NODES, DIM, DIM, 1.0f, loc_kb + l*DIM,
             nullptr, 0, k16, DIM, nullptr, 0);
        gemm(a16, DIM, wT[l*4+2], DIM, N_NODES, DIM, DIM, 1.0f, loc_vb + l*DIM,
             nullptr, 0, nullptr, 0, vt16, N_NODES);
        for (int hh = 0; hh < HEADS; ++hh) {
            // S = (Q_h K_h^T) / sqrt(DH)
            gemm(q16 + hh*DHEAD, DIM, k16 + hh*DHEAD, DIM,
                 N_NODES, N_NODES, DHEAD, inv_sqrt_dh, nullptr,
                 S, N_NODES, nullptr, 0, nullptr, 0);
            // += edge-type bias at graph edges
            hipLaunchKernelGGL(scatter_edge_bias, dim3((EDGES + 255) / 256), B256,
                               0, stream, S, ei, et, loc_eb + l*NET_T*HEADS, hh);
            // same-community masked softmax -> P16
            hipLaunchKernelGGL(masked_softmax, dim3(N_NODES), B256, 0, stream,
                               S, P16, cid, 0);
            // O_h = P @ V_h   (uses V^T rows, K-dim = N)
            gemm(P16, N_NODES, vt16 + (size_t)hh*DHEAD*N_NODES, N_NODES,
                 N_NODES, DHEAD, N_NODES, 1.0f, nullptr,
                 tmp2 + hh*DHEAD, DIM, nullptr, 0, nullptr, 0);
        }
        // o-projection + residual + LN
        hipLaunchKernelGGL(cvt_f16_kernel, dim3((int)(ND / 256)), B256, 0, stream,
                           tmp2, nullptr, att16, (int)ND);
        gemm(att16, DIM, wT[l*4+3], DIM, N_NODES, DIM, DIM, 1.0f, loc_ob + l*DIM,
             tmp, DIM, nullptr, 0, nullptr, 0);
        hipLaunchKernelGGL(residual_layernorm, dim3(N_NODES), B256, 0, stream,
                           h, tmp, loc_ln_g + l*DIM, loc_ln_b + l*DIM);
    }

    // ---- global interaction layers ----
    for (int g = 0; g < L_GLB; ++g) {
        hipLaunchKernelGGL(cvt_f16_kernel, dim3((int)(ND / 256)), B256, 0, stream,
                           h, nullptr, a16, (int)ND);
        gemm(a16, DIM, wT[8+g*4+0], DIM, N_NODES, DIM, DIM, 1.0f, glob_qb + g*DIM,
             nullptr, 0, q16, DIM, nullptr, 0);
        gemm(a16, DIM, wT[8+g*4+1], DIM, N_NODES, DIM, DIM, 1.0f, glob_kb + g*DIM,
             nullptr, 0, k16, DIM, nullptr, 0);
        gemm(a16, DIM, wT[8+g*4+2], DIM, N_NODES, DIM, DIM, 1.0f, glob_vb + g*DIM,
             nullptr, 0, nullptr, 0, vt16, N_NODES);
        // S = Q K^T / sqrt(DH)  (full D, single head)
        gemm(q16, DIM, k16, DIM, N_NODES, N_NODES, DIM, inv_sqrt_dh, nullptr,
             S, N_NODES, nullptr, 0, nullptr, 0);
        // different-community masked softmax
        hipLaunchKernelGGL(masked_softmax, dim3(N_NODES), B256, 0, stream,
                           S, P16, cid, 1);
        // attn @ V
        gemm(P16, N_NODES, vt16, N_NODES, N_NODES, DIM, N_NODES, 1.0f, nullptr,
             tmp2, DIM, nullptr, 0, nullptr, 0);
        hipLaunchKernelGGL(cvt_f16_kernel, dim3((int)(ND / 256)), B256, 0, stream,
                           tmp2, nullptr, att16, (int)ND);
        gemm(att16, DIM, wT[8+g*4+3], DIM, N_NODES, DIM, DIM, 1.0f, glob_ob + g*DIM,
             tmp, DIM, nullptr, 0, nullptr, 0);
        hipLaunchKernelGGL(residual_layernorm, dim3(N_NODES), B256, 0, stream,
                           h, tmp, glob_ln_g + g*DIM, glob_ln_b + g*DIM);
    }

    // ---- output projection: out = h @ out_w + out_b ----
    hipLaunchKernelGGL(cvt_f16_kernel, dim3((int)(ND / 256)), B256, 0, stream,
                       h, nullptr, a16, (int)ND);
    gemm(a16, DIM, woutT, DIM, N_NODES, OUT_D, DIM, 1.0f, out_b,
         out, OUT_D, nullptr, 0, nullptr, 0);
}
